// LSTMModel_26328149525217
// MI455X (gfx1250) — matrix operand order repaired and checked by reading, not compiled, not run on hardware
//
#include <hip/hip_runtime.h>
#include <hip/hip_bf16.h>

typedef __attribute__((ext_vector_type(16))) _Float16     v16h;
typedef __attribute__((ext_vector_type(8)))  float        v8f;
typedef __attribute__((ext_vector_type(4)))  unsigned int v4u;

namespace {
constexpr int kB = 2048;   // batch
constexpr int kT = 512;    // time steps
constexpr int kD = 16;     // input feature dim
constexpr int kH = 48;     // hidden dim
constexpr int kG = 192;    // 4*kH gate columns
constexpr int BT = 16;     // batch rows per workgroup (one 16-row M tile)
constexpr int K0 = 64;     // layer0 concat K: [x(16) | h0(48)]  -> 2 x K32
constexpr int K1 = 96;     // layer1 concat K: [h0(48) | h1(48)] -> 3 x K32
constexpr int NTH = 384;   // 12 waves, one per 16-col N tile of the 192 gates
}

// Raw v_rcp_f32: keeps the serial recurrence chain short.
__device__ __forceinline__ float fast_rcp(float x) {
  return __builtin_amdgcn_rcpf(x);
}
__device__ __forceinline__ float fsigmoid(float x) {
  return fast_rcp(1.0f + __expf(-x));
}
__device__ __forceinline__ float ftanh(float x) {
  // 1 - 2/(e^{2x}+1): stable at +/-inf
  return 1.0f - 2.0f * fast_rcp(__expf(2.0f * x) + 1.0f);
}

__device__ __forceinline__ v8f wmma16x16x32(v16h a, v16h b, v8f c) {
  return __builtin_amdgcn_wmma_f32_16x16x32_f16(false, a, false, b, (short)0, c,
                                                false, false);
}

// Two 16B LDS loads -> one 16-bit A fragment (p already points at this lane's
// K-chunk base; halves 0..7 at p, halves 8..15 at p+16).
__device__ __forceinline__ v16h load_frag_at(const _Float16* p) {
  union { v16h h; v4u u[2]; } f;
  f.u[0] = *reinterpret_cast<const v4u*>(p);
  f.u[1] = *reinterpret_cast<const v4u*>(p + 16);
  return f.h;
}

// Build a 16-bit B-matrix 32x16 fragment (one kc chunk) straight from the
// f32 weights in global memory (done once, held in VGPRs for all 512 steps).
// Lane L: N = ntile*16 + L%16; halves i hold K = kc*32 + 8*(L/16) + i (i<8),
//   kc*32 + 16 + 8*(L/16) + (i-8) (i>=8).
//   k <  DIN : W_ih[n][k] ;  k >= DIN : W_hh[n][k-DIN]
template <int DIN>
__device__ __forceinline__ v16h make_b_frag(const float* __restrict__ Wih,
                                            const float* __restrict__ Whh,
                                            int ntile, int kc, int lane) {
  const int n  = ntile * 16 + (lane & 15);
  const int hi = lane >> 4;
  v16h b;
#pragma unroll
  for (int i = 0; i < 16; ++i) {
    const int k = kc * 32 + (i < 8 ? hi * 8 + i : 16 + hi * 8 + (i - 8));
    const float w = (k < DIN) ? Wih[n * DIN + k] : Whh[n * kH + (k - DIN)];
    b[i] = (_Float16)w;
  }
  return b;
}

__device__ __forceinline__ void store_gates(float* gst, v8f acc) {
#pragma unroll
  for (int r = 0; r < 8; ++r) gst[r * kG] = acc[r];
}

__global__ __launch_bounds__(NTH, 1) void lstm2_fused_kernel(
    const float* __restrict__ x,
    const float* __restrict__ Wih0, const float* __restrict__ Whh0,
    const float* __restrict__ bih0, const float* __restrict__ bhh0,
    const float* __restrict__ Wih1, const float* __restrict__ Whh1,
    const float* __restrict__ bih1, const float* __restrict__ bhh1,
    const float* __restrict__ fcw, const float* __restrict__ fcb,
    float* __restrict__ out) {
  // LDS (all 16B aligned). ~35 KB total.
  __shared__ __align__(16) _Float16 sA0[BT * K0];  // [16][64]: [x_t | h0]
  __shared__ __align__(16) _Float16 sA1[BT * K1];  // [16][96]: [h0_t | h1]
  __shared__ __align__(16) float    sG0[BT * kG];  // layer0 gate pre-acts
  __shared__ __align__(16) float    sG1[BT * kG];  // layer1 gate pre-acts
  __shared__ __align__(16) float    sC0[BT * kH];  // layer0 cell state
  __shared__ __align__(16) float    sC1[BT * kH];  // layer1 cell state

  const int tid   = threadIdx.x;
  const int lane  = tid & 31;
  const int ntile = tid >> 5;              // 0..11: which 16 gate columns
  const int brow  = blockIdx.x * BT;       // global batch base row

  // ---- one-time: weight fragments into VGPRs (reused for all steps) ----
  const v16h b0f0 = make_b_frag<kD>(Wih0, Whh0, ntile, 0, lane);
  const v16h b0f1 = make_b_frag<kD>(Wih0, Whh0, ntile, 1, lane);
  const v16h b1f0 = make_b_frag<kH>(Wih1, Whh1, ntile, 0, lane);
  const v16h b1f1 = make_b_frag<kH>(Wih1, Whh1, ntile, 1, lane);
  const v16h b1f2 = make_b_frag<kH>(Wih1, Whh1, ntile, 2, lane);

  const int coln   = ntile * 16 + (lane & 15);     // gate column of C/D frag
  const float bias0 = bih0[coln] + bhh0[coln];
  const float bias1 = bih1[coln] + bhh1[coln];

  // ---- loop-invariant LDS addressing, computed once ----
  // A-fragment bases: lane L -> row = L%16, halves offset 8*(L/16)
  const _Float16* a0base = sA0 + (lane & 15) * K0 + (lane >> 4) * 8;
  const _Float16* a1base = sA1 + (lane & 15) * K1 + (lane >> 4) * 8;
  // C/D gate-store base: rows rb..rb+7 of column coln
  float* g0st = sG0 + ((lane >> 4) * 8) * kG + coln;
  float* g1st = sG1 + ((lane >> 4) * 8) * kG + coln;

  // Pointwise quad decode (2 quads per lane, loop-invariant offsets)
  int gofs[2], cofs[2], a0ofs[2], a1ofs[2], a1hofs[2];
#pragma unroll
  for (int q = 0; q < 2; ++q) {
    const int idx = tid + q * NTH;          // 0..767 over 16x48 state tile
    const int m = idx / kH, hc = idx - m * kH;
    gofs[q]   = m * kG + hc;                // +0/48/96/144 for i/f/g/o
    cofs[q]   = m * kH + hc;
    a0ofs[q]  = m * K0 + kD + hc;           // h0 -> layer0 recurrent slot
    a1ofs[q]  = m * K1 + hc;                // h0 -> layer1 input slot
    a1hofs[q] = m * K1 + kH + hc;           // h1 -> layer1 recurrent slot
  }

  // x staging: one f32 per thread (first 256 threads); step stride = kD
  const int xm = tid >> 4, xk = tid & 15;
  const bool xthr = tid < BT * kD;
  const float* xp = x + (size_t)(brow + xm) * kT * kD + xk;
  _Float16* xst = sA0 + xm * K0 + xk;

  // one LSTM cell update for quad q; returns h (f32)
  auto cell = [&](const float* __restrict__ sGp, float* __restrict__ sCp,
                  int q) -> float {
    const int g = gofs[q];
    const float ig = fsigmoid(sGp[g]);
    const float fg = fsigmoid(sGp[g + kH]);
    const float gg = ftanh   (sGp[g + 2 * kH]);
    const float og = fsigmoid(sGp[g + 3 * kH]);
    const float c  = fg * sCp[cofs[q]] + ig * gg;
    sCp[cofs[q]] = c;
    return og * ftanh(c);
  };

  // ---- zero recurrent state, stage x_0 ----
  for (int i = tid; i < BT * kH; i += NTH) { sC0[i] = 0.0f; sC1[i] = 0.0f; }
  for (int i = tid; i < BT * K0; i += NTH) sA0[i] = (_Float16)0.0f;
  for (int i = tid; i < BT * K1; i += NTH) sA1[i] = (_Float16)0.0f;
  if (xthr) *xst = (_Float16)xp[0];
  __syncthreads();

  // ---- prologue: gates0(0), pw0(0) ----
  {
    const float xv = xthr ? xp[kD] : 0.0f;  // x_1
    v8f acc0 = {bias0, bias0, bias0, bias0, bias0, bias0, bias0, bias0};
    acc0 = wmma16x16x32(load_frag_at(a0base), b0f0, acc0);
    acc0 = wmma16x16x32(load_frag_at(a0base + 32), b0f1, acc0);
    store_gates(g0st, acc0);
    __syncthreads();
#pragma unroll
    for (int q = 0; q < 2; ++q) {
      const _Float16 h = (_Float16)cell(sG0, sC0, q);
      sA0[a0ofs[q]] = h;
      sA1[a1ofs[q]] = h;
    }
    if (xthr) *xst = (_Float16)xv;
    __syncthreads();
  }

  // ---- skewed steady state ----
  // iteration t (0..510): WMMA{gates1(t), gates0(t+1)} ; PW{pw1(t), pw0(t+1)}
  for (int t = 0; t < kT - 1; ++t) {
    // prefetch x_{t+2} (overlaps the WMMA phase)
    const bool xld = xthr && (t + 2 < kT);
    float xv = 0.0f;
    if (xld) xv = xp[(size_t)(t + 2) * kD];

    // merged WMMA phase: two independent accumulator chains, constant-offset
    // ds_load_b128 bursts
    const v16h a10 = load_frag_at(a1base);
    const v16h a11 = load_frag_at(a1base + 32);
    const v16h a12 = load_frag_at(a1base + 64);
    const v16h a00 = load_frag_at(a0base);
    const v16h a01 = load_frag_at(a0base + 32);
    v8f acc1 = {bias1, bias1, bias1, bias1, bias1, bias1, bias1, bias1};
    v8f acc0 = {bias0, bias0, bias0, bias0, bias0, bias0, bias0, bias0};
    acc1 = wmma16x16x32(a10, b1f0, acc1);
    acc0 = wmma16x16x32(a00, b0f0, acc0);
    acc1 = wmma16x16x32(a11, b1f1, acc1);
    acc0 = wmma16x16x32(a01, b0f1, acc0);
    acc1 = wmma16x16x32(a12, b1f2, acc1);
    store_gates(g1st, acc1);
    store_gates(g0st, acc0);
    __syncthreads();

    // merged pointwise phase: pw1(t) and pw0(t+1) are independent
#pragma unroll
    for (int q = 0; q < 2; ++q)
      sA1[a1hofs[q]] = (_Float16)cell(sG1, sC1, q);   // h1_t
#pragma unroll
    for (int q = 0; q < 2; ++q) {
      const _Float16 h = (_Float16)cell(sG0, sC0, q); // h0_{t+1}
      sA0[a0ofs[q]] = h;
      sA1[a1ofs[q]] = h;
    }
    if (xld) *xst = (_Float16)xv;                     // x_{t+2}
    __syncthreads();
  }

  // ---- epilogue: gates1(T-1), pw1(T-1) ----
  {
    v8f acc1 = {bias1, bias1, bias1, bias1, bias1, bias1, bias1, bias1};
    acc1 = wmma16x16x32(load_frag_at(a1base), b1f0, acc1);
    acc1 = wmma16x16x32(load_frag_at(a1base + 32), b1f1, acc1);
    acc1 = wmma16x16x32(load_frag_at(a1base + 64), b1f2, acc1);
    store_gates(g1st, acc1);
    __syncthreads();
#pragma unroll
    for (int q = 0; q < 2; ++q)
      sA1[a1hofs[q]] = (_Float16)cell(sG1, sC1, q);
    __syncthreads();
  }

  // final FC: logits[b] = h1_T . fc_w + fc_b ; out = sigmoid(logits)
  if (tid < BT) {
    float acc = fcb[0];
#pragma unroll
    for (int n = 0; n < kH; ++n)
      acc += (float)sA1[tid * K1 + kH + n] * fcw[n];
    out[brow + tid] = fsigmoid(acc);
  }
}

extern "C" void kernel_launch(void* const* d_in, const int* in_sizes, int n_in,
                              void* d_out, int out_size, void* d_ws,
                              size_t ws_size, hipStream_t stream) {
  (void)in_sizes; (void)n_in; (void)out_size; (void)d_ws; (void)ws_size;
  const float* x    = (const float*)d_in[0];
  const float* Wih0 = (const float*)d_in[1];
  const float* Whh0 = (const float*)d_in[2];
  const float* bih0 = (const float*)d_in[3];
  const float* bhh0 = (const float*)d_in[4];
  const float* Wih1 = (const float*)d_in[5];
  const float* Whh1 = (const float*)d_in[6];
  const float* bih1 = (const float*)d_in[7];
  const float* bhh1 = (const float*)d_in[8];
  const float* fcw  = (const float*)d_in[9];
  const float* fcb  = (const float*)d_in[10];
  float* out = (float*)d_out;

  dim3 grid(kB / BT);   // 128 workgroups, each owns 16 batch rows end-to-end
  dim3 block(NTH);      // 12 waves
  hipLaunchKernelGGL(lstm2_fused_kernel, grid, block, 0, stream,
                     x, Wih0, Whh0, bih0, bhh0, Wih1, Whh1, bih1, bhh1,
                     fcw, fcb, out);
}
